// Net_12111807775380
// MI455X (gfx1250) — matrix operand order repaired and checked by reading, not compiled, not run on hardware
//
#include <hip/hip_runtime.h>
#include <hip/hip_bf16.h>

typedef __attribute__((ext_vector_type(16))) __bf16 v16bf;
typedef __attribute__((ext_vector_type(8)))  __bf16 v8bf;
typedef __attribute__((ext_vector_type(8)))  float  v8f;

#define N_NODES 50000
#define N_EDGES 800000
#define LOG_2PI 1.8378770664093453f

// ---------------------------------------------------------------------------
// Pack a row-major f32 weight W[K,N] into the CDNA5 WMMA B-operand lane layout
// (bf16). Layout: Bp[((kt*NT_TOTAL + ntg)*32 + lane)*16 + e] where within a
// 32-K block:  lane = ((k>>3)&1) * 16 + (n&15),  e = (k&7) + ((k&31)>>4)*8,  ntg = ntg_off+(n>>4)
// ---------------------------------------------------------------------------
__global__ void k_pack_w(const float* __restrict__ W, __bf16* __restrict__ out,
                         int K, int N, int ntg_off, int NT_TOTAL) {
  int idx = blockIdx.x * 256 + threadIdx.x;
  if (idx >= K * N) return;
  int k = idx / N, n = idx % N;
  int kt  = k >> 5;
  int kr  = k & 31;
  int lhi = (kr >> 3) & 1;
  int e   = (kr & 7) + ((kr >> 4) << 3);
  int lane = lhi * 16 + (n & 15);
  int ntg  = ntg_off + (n >> 4);
  out[((size_t)(kt * NT_TOTAL + ntg) * 32 + lane) * 16 + e] = (__bf16)W[idx];
}

// ---------------------------------------------------------------------------
// Edge-parallel scatter-sum:  out[dst[e], f] += X[src[e], f] * (a ? a[e,f] : 1)
// ---------------------------------------------------------------------------
__global__ void k_scatter(const float* __restrict__ X, const float* __restrict__ Agate,
                          const int* __restrict__ src, const int* __restrict__ dst,
                          float* __restrict__ out) {
  long idx = (long)blockIdx.x * 256 + threadIdx.x;
  if (idx >= (long)N_EDGES * 64) return;
  long e = idx >> 6;
  int  f = (int)(idx & 63);
  float v = X[(size_t)src[e] * 64 + f];
  if (Agate) v *= Agate[idx];
  unsafeAtomicAdd(&out[(size_t)dst[e] * 64 + f], v);
}

// ---------------------------------------------------------------------------
// Node GEMM: out = act(A[M,64] @ W[64,64] + bias), W pre-packed bf16 (NT=4).
// One wave = one 16x16 tile; block of 128 threads covers a full M-tile row.
// ---------------------------------------------------------------------------
template <bool RELU, bool TOBF>
__global__ void k_gemm64(const float* __restrict__ A, const __bf16* __restrict__ Bp,
                         const float* __restrict__ bias, float* __restrict__ outf,
                         __bf16* __restrict__ outb) {
  int nt   = threadIdx.x >> 5;     // 4 waves -> 4 N-tiles
  int lane = threadIdx.x & 31;
  int row0 = blockIdx.x * 16;
  int m    = lane & 15;
  int hv   = lane >> 4;
  const float* arow = A + (size_t)(row0 + m) * 64;

  v16bf a0, a1;
#pragma unroll
  for (int kb = 0; kb < 2; ++kb) {
    v16bf av;
    int base = kb * 32 + hv * 8;
#pragma unroll
    for (int j = 0; j < 8; ++j) av[j] = (__bf16)arow[base + j];
#pragma unroll
    for (int j = 0; j < 8; ++j) av[8 + j] = (__bf16)arow[base + 16 + j];
    if (kb == 0) a0 = av; else a1 = av;
  }

  const v16bf* b0 = (const v16bf*)(Bp + ((size_t)(0 * 4 + nt) * 32 + lane) * 16);
  const v16bf* b1 = (const v16bf*)(Bp + ((size_t)(1 * 4 + nt) * 32 + lane) * 16);
  v8f c = {};
  c = __builtin_amdgcn_wmma_f32_16x16x32_bf16(false, a0, false, *b0, (short)0, c, false, false);
  c = __builtin_amdgcn_wmma_f32_16x16x32_bf16(false, a1, false, *b1, (short)0, c, false, false);

  int col = nt * 16 + (lane & 15);
  float bv = bias[col];
#pragma unroll
  for (int r = 0; r < 8; ++r) {
    int rr = row0 + r + 8 * hv;
    float v = c[r] + bv;
    if (RELU) v = fmaxf(v, 0.0f);
    if (TOBF) outb[(size_t)rr * 64 + col] = (__bf16)v;
    else      outf[(size_t)rr * 64 + col] = v;
  }
}

// ---------------------------------------------------------------------------
// Fused edge MLP: zz = [z[src], z[dst]] (K=128, bf16), 4 heads fused as N=256.
// One wave = 16 edges; A kept resident (4 K-blocks, 32 VGPRs); weights staged
// in LDS (64 KB). For the a_first heads the gated scatter into t0 is fused
// (a_first never touches HBM); a_rest is materialized (consumed twice later).
// NLL partials accumulated via wave reduction + 2 atomics.
// ---------------------------------------------------------------------------
__global__ void k_edge_mlp(const __bf16* __restrict__ Z, const int* __restrict__ src,
                           const int* __restrict__ dst, const __bf16* __restrict__ Bp,
                           const float* __restrict__ b_mu_f, const float* __restrict__ b_ls_f,
                           const float* __restrict__ b_mu,  const float* __restrict__ b_ls,
                           const float* __restrict__ eps_f, const float* __restrict__ eps_r,
                           const float* __restrict__ x, float* __restrict__ t0,
                           float* __restrict__ a_rest, float* __restrict__ nll) {
  __shared__ __bf16 sB[32768];                       // 64 KB: 4 kt x 16 ntg x 32 x 16
  {
    const uint4* g = (const uint4*)Bp;
    uint4* l = (uint4*)sB;
    for (int i = threadIdx.x; i < 4096; i += 256) l[i] = g[i];
  }
  __syncthreads();

  int wave = threadIdx.x >> 5, lane = threadIdx.x & 31;
  long tile = (long)blockIdx.x * 8 + wave;
  long e0 = tile * 16;
  int  m  = lane & 15, hv = lane >> 4;
  long edge = e0 + m;
  int si = src[edge], di = dst[edge];
  const __bf16* zs = Z + (size_t)si * 64;
  const __bf16* zd = Z + (size_t)di * 64;

  v16bf A[4];
#pragma unroll
  for (int kb = 0; kb < 4; ++kb) {
#pragma unroll
    for (int hh = 0; hh < 2; ++hh) {
      int Kst = kb * 32 + hh * 16 + hv * 8;          // multiple of 8 -> 16B load
      const __bf16* ptr = (Kst < 64) ? (zs + Kst) : (zd + (Kst - 64));
      v8bf v = *(const v8bf*)ptr;
#pragma unroll
      for (int j = 0; j < 8; ++j) A[kb][hh * 8 + j] = v[j];
    }
  }

  float nllA = 0.0f, nllB = 0.0f;
#pragma unroll
  for (int p = 0; p < 8; ++p) {
    const bool first = (p < 4);
    const int q = p & 3;
    const int ntg_mu = first ? q     : 8 + q;
    const int ntg_ls = first ? 4 + q : 12 + q;
    v8f cmu = {}, cls = {};
#pragma unroll
    for (int kb = 0; kb < 4; ++kb) {
      const v16bf* bmu = (const v16bf*)(sB + ((kb * 16 + ntg_mu) * 32 + lane) * 16);
      const v16bf* bls = (const v16bf*)(sB + ((kb * 16 + ntg_ls) * 32 + lane) * 16);
      cmu = __builtin_amdgcn_wmma_f32_16x16x32_bf16(false, A[kb], false, *bmu, (short)0, cmu, false, false);
      cls = __builtin_amdgcn_wmma_f32_16x16x32_bf16(false, A[kb], false, *bls, (short)0, cls, false, false);
    }
    int col = q * 16 + (lane & 15);
    float bmuv = first ? b_mu_f[col] : b_mu[col];
    float blsv = first ? b_ls_f[col] : b_ls[col];
    const float* eps = first ? eps_f : eps_r;
#pragma unroll
    for (int r = 0; r < 8; ++r) {
      long er = e0 + r + 8 * hv;
      float muv = cmu[r] + bmuv;
      float sg  = __expf(cls[r] + blsv);
      float av  = muv + sg * eps[er * 64 + col];
      float dd  = av - 1.0f;
      if (first) {
        // fused gated scatter: t0[dst[er], col] += x[src[er], col] * av
        int si_r = __shfl(si, r + 8 * hv, 32);
        int di_r = __shfl(di, r + 8 * hv, 32);
        float xv = x[(size_t)si_r * 64 + col];
        unsafeAtomicAdd(&t0[(size_t)di_r * 64 + col], xv * av);
        nllA += 0.5f * dd * dd;
      } else {
        a_rest[er * 64 + col] = av;
        nllB += 0.5f * dd * dd;
      }
    }
  }
  for (int off = 16; off; off >>= 1) {
    nllA += __shfl_xor(nllA, off, 32);
    nllB += __shfl_xor(nllB, off, 32);
  }
  if (lane == 0) {
    unsafeAtomicAdd(&nll[0], nllA);
    unsafeAtomicAdd(&nll[1], nllB);
  }
}

// ---------------------------------------------------------------------------
// Column sum of T[N_NODES,64] (for g = colsum(t2) @ W2 + N*b2; last gconv has
// no activation so h3 never needs to be materialized).
// ---------------------------------------------------------------------------
__global__ void k_colsum(const float* __restrict__ T, float* __restrict__ cs) {
  int f  = threadIdx.x;                 // 64 threads
  int r0 = blockIdx.x * 256;
  int r1 = r0 + 256; if (r1 > N_NODES) r1 = N_NODES;
  float s = 0.0f;
  for (int r = r0; r < r1; ++r) s += T[(size_t)r * 64 + f];
  unsafeAtomicAdd(&cs[f], s);
}

// ---------------------------------------------------------------------------
// Readout head + NLL finalize. out[0..31] = head, out[32] = nll_reg.
// ---------------------------------------------------------------------------
__global__ void k_final(const float* __restrict__ cs,
                        const float* __restrict__ W2,  const float* __restrict__ b2,
                        const float* __restrict__ Wd1, const float* __restrict__ bd1,
                        const float* __restrict__ Wd2, const float* __restrict__ bd2,
                        const float* __restrict__ nll, float* __restrict__ out) {
  __shared__ float sg[64], sh[64];
  int t = threadIdx.x;                  // 64 threads
  float acc = (float)N_NODES * b2[t];
  for (int k = 0; k < 64; ++k) acc += cs[k] * W2[k * 64 + t];
  sg[t] = fmaxf(acc, 0.0f);             // relu(g)
  __syncthreads();
  float acc2 = bd1[t];
  for (int k = 0; k < 64; ++k) acc2 += sg[k] * Wd1[k * 64 + t];
  sh[t] = fmaxf(acc2, 0.0f);
  __syncthreads();
  if (t < 32) {
    float acc3 = bd2[t];
    for (int k = 0; k < 64; ++k) acc3 += sh[k] * Wd2[k * 32 + t];
    out[t] = acc3;
  }
  if (t == 0)
    out[32] = (nll[0] + nll[1]) * (1.0f / ((float)N_EDGES * 64.0f)) + LOG_2PI;
}

// ---------------------------------------------------------------------------
extern "C" void kernel_launch(void* const* d_in, const int* in_sizes, int n_in,
                              void* d_out, int out_size, void* d_ws, size_t ws_size,
                              hipStream_t stream) {
  const float* x      = (const float*)d_in[0];
  const int*   src    = (const int*)d_in[1];
  const int*   dst    = (const int*)d_in[2];
  const float* eps_f  = (const float*)d_in[3];
  const float* eps_r  = (const float*)d_in[4];
  const float* W_enc0 = (const float*)d_in[5];  const float* b_enc0 = (const float*)d_in[6];
  const float* W_enc1 = (const float*)d_in[7];  const float* b_enc1 = (const float*)d_in[8];
  const float* W_mu_f = (const float*)d_in[9];  const float* b_mu_f = (const float*)d_in[10];
  const float* W_ls_f = (const float*)d_in[11]; const float* b_ls_f = (const float*)d_in[12];
  const float* W_mu   = (const float*)d_in[13]; const float* b_mu   = (const float*)d_in[14];
  const float* W_ls   = (const float*)d_in[15]; const float* b_ls   = (const float*)d_in[16];
  const float* W0     = (const float*)d_in[17]; const float* b0     = (const float*)d_in[18];
  const float* W1     = (const float*)d_in[19]; const float* b1     = (const float*)d_in[20];
  const float* W2     = (const float*)d_in[21]; const float* b2     = (const float*)d_in[22];
  const float* Wd1    = (const float*)d_in[23]; const float* bd1    = (const float*)d_in[24];
  const float* Wd2    = (const float*)d_in[25]; const float* bd2    = (const float*)d_in[26];

  char* ws = (char*)d_ws;
  size_t off = 0;
  auto alloc = [&](size_t bytes) -> void* {
    void* p = ws + off;
    off += (bytes + 255) & ~(size_t)255;
    return p;
  };
  const size_t NODE_BYTES = (size_t)N_NODES * 64 * sizeof(float);
  float*  nA      = (float*)alloc(NODE_BYTES);
  float*  nB      = (float*)alloc(NODE_BYTES);
  float*  nC      = (float*)alloc(NODE_BYTES);
  __bf16* zbf     = (__bf16*)alloc((size_t)N_NODES * 64 * sizeof(__bf16));
  float*  a_rest  = (float*)alloc((size_t)N_EDGES * 64 * sizeof(float));
  __bf16* pw_enc0 = (__bf16*)alloc(4096 * sizeof(__bf16));
  __bf16* pw_enc1 = (__bf16*)alloc(4096 * sizeof(__bf16));
  __bf16* pwW0    = (__bf16*)alloc(4096 * sizeof(__bf16));
  __bf16* pwW1    = (__bf16*)alloc(4096 * sizeof(__bf16));
  __bf16* pwe     = (__bf16*)alloc(32768 * sizeof(__bf16));
  float*  nllbuf  = (float*)alloc(2 * sizeof(float));
  float*  cs      = (float*)alloc(64 * sizeof(float));

  const int EG = (int)(((long)N_EDGES * 64 + 255) / 256);  // scatter grid
  const int MT = N_NODES / 16;                             // 3125 M-tiles
  const int ET = N_EDGES / 16 / 8;                         // 6250 blocks (8 waves)

  // --- weight packing (tiny, L2-resident) ---
  k_pack_w<<<16, 256, 0, stream>>>(W_enc0, pw_enc0, 64, 64, 0, 4);
  k_pack_w<<<16, 256, 0, stream>>>(W_enc1, pw_enc1, 64, 64, 0, 4);
  k_pack_w<<<16, 256, 0, stream>>>(W0,     pwW0,    64, 64, 0, 4);
  k_pack_w<<<16, 256, 0, stream>>>(W1,     pwW1,    64, 64, 0, 4);
  k_pack_w<<<32, 256, 0, stream>>>(W_mu_f, pwe, 128, 64, 0,  16);
  k_pack_w<<<32, 256, 0, stream>>>(W_ls_f, pwe, 128, 64, 4,  16);
  k_pack_w<<<32, 256, 0, stream>>>(W_mu,   pwe, 128, 64, 8,  16);
  k_pack_w<<<32, 256, 0, stream>>>(W_ls,   pwe, 128, 64, 12, 16);

  // --- encoder gconv 1: s1 = segsum(x[src]); z1 = relu(s1@Wenc0+b) ---
  hipMemsetAsync(nA, 0, NODE_BYTES, stream);
  k_scatter<<<EG, 256, 0, stream>>>(x, nullptr, src, dst, nA);
  k_gemm64<true, false><<<MT, 128, 0, stream>>>(nA, pw_enc0, b_enc0, nB, nullptr);
  // --- encoder gconv 2 -> z (bf16) ---
  hipMemsetAsync(nC, 0, NODE_BYTES, stream);
  k_scatter<<<EG, 256, 0, stream>>>(nB, nullptr, src, dst, nC);
  k_gemm64<true, true><<<MT, 128, 0, stream>>>(nC, pw_enc1, b_enc1, nullptr, zbf);

  // --- fused edge MLP + reparam + NLL + fused a_first gated scatter -> t0 ---
  hipMemsetAsync(nllbuf, 0, 2 * sizeof(float), stream);
  hipMemsetAsync(nA, 0, NODE_BYTES, stream);               // t0
  k_edge_mlp<<<ET, 256, 0, stream>>>(zbf, src, dst, pwe, b_mu_f, b_ls_f, b_mu, b_ls,
                                     eps_f, eps_r, x, nA, a_rest, nllbuf);

  // --- gated gconv stack ---
  k_gemm64<true, false><<<MT, 128, 0, stream>>>(nA, pwW0, b0, nB, nullptr);   // h1
  hipMemsetAsync(nC, 0, NODE_BYTES, stream);
  k_scatter<<<EG, 256, 0, stream>>>(nB, a_rest, src, dst, nC);
  k_gemm64<true, false><<<MT, 128, 0, stream>>>(nC, pwW1, b1, nA, nullptr);   // h2
  hipMemsetAsync(nB, 0, NODE_BYTES, stream);
  k_scatter<<<EG, 256, 0, stream>>>(nA, a_rest, src, dst, nB);                // t2

  // --- readout: g = colsum(t2)@W2 + N*b2 ; head ; nll finalize ---
  hipMemsetAsync(cs, 0, 64 * sizeof(float), stream);
  k_colsum<<<(N_NODES + 255) / 256, 64, 0, stream>>>(nB, cs);
  k_final<<<1, 64, 0, stream>>>(cs, W2, b2, Wd1, bd1, Wd2, bd2, nllbuf, (float*)d_out);
}
